// Attention_87497073754296
// MI455X (gfx1250) — matrix operand order repaired and checked by reading, not compiled, not run on hardware
//
#include <hip/hip_runtime.h>

// ---------------------------------------------------------------------------
// Attention: Z = softmax((Y W^T)(Y W^T)^T) Y      N=4096 tokens, D=1024 dims
// Flash-attention style fusion on CDNA5 wave32 WMMA (v_wmma_f32_16x16x32_f16).
// Workspace layout (18 MB total, all L2-resident on MI455X's 192MB L2):
//   [0,      2MB)  Wh : f16 copy of W_param            (1024x1024)
//   [2MB,   10MB)  Yt : f16 transpose of Y             (1024x4096)
//   [10MB,  18MB)  G  : f16 G = Y @ W^T                (4096x1024)
// ---------------------------------------------------------------------------

typedef __attribute__((ext_vector_type(16))) _Float16 v16h;
typedef __attribute__((ext_vector_type(8)))  _Float16 v8h;
typedef __attribute__((ext_vector_type(8)))  float    v8f;
typedef __attribute__((ext_vector_type(8)))  float    vf8;

constexpr int NT = 4096;   // tokens
constexpr int DD = 1024;   // feature dim

static __device__ __forceinline__ v8f wmma16(v16h a, v16h b, v8f c) {
  // D = A(16x32 f16) x B(32x16 f16) + C(16x16 f32)
  return __builtin_amdgcn_wmma_f32_16x16x32_f16(
      /*neg_a=*/false, a, /*neg_b=*/false, b,
      /*c_mod=*/(short)0, c, /*reuse_a=*/false, /*reuse_b=*/false);
}

static __device__ __forceinline__ v16h cat8(v8h a, v8h b) {
  return __builtin_shufflevector(a, b, 0,1,2,3,4,5,6,7,8,9,10,11,12,13,14,15);
}

// reduce across the 16-lane groups (lanes 0-15 / 16-31 reduce independently)
static __device__ __forceinline__ float redmax16(float v) {
  #pragma unroll
  for (int m = 8; m >= 1; m >>= 1) v = fmaxf(v, __shfl_xor(v, m, 32));
  return v;
}
static __device__ __forceinline__ float redsum16(float v) {
  #pragma unroll
  for (int m = 8; m >= 1; m >>= 1) v += __shfl_xor(v, m, 32);
  return v;
}

// ------------------------- staging kernels ---------------------------------

__global__ void k_cast_w(const float* __restrict__ W, _Float16* __restrict__ Wh) {
  int i = blockIdx.x * 256 + threadIdx.x;          // over DD*DD
  Wh[i] = (_Float16)W[i];
}

__global__ void k_transpose_y(const float* __restrict__ Y, _Float16* __restrict__ Yt) {
  int i = blockIdx.x * 256 + threadIdx.x;          // i = d*NT + n
  int d = i >> 12;                                 // NT = 4096
  int n = i & (NT - 1);
  Yt[i] = (_Float16)Y[n * DD + d];
}

// ------------------------- GEMM1: G = Y @ W^T ------------------------------
// 8 waves/block; each wave owns a 16x128 output tile. K loop steps of 32.
// A-frag (16x32 f16): lane<16 holds row m=lane, K {k..k+15};
//                     lane>=16 same row, K {k+16..k+31}.
// B-frag (32x16 f16): B[k][n] = W[n][k]; lane holds col n=lane&15,
//                     16 contiguous halves of row n at k + 16*(lane>>4).
// All 8 B-fragments are preloaded before the WMMA chain so the memory clause
// overlaps with the matrix pipe instead of s_wait_loadcnt 0 per WMMA.

__launch_bounds__(256)
__global__ void k_gemm1(const float* __restrict__ Y, const _Float16* __restrict__ Wh,
                        _Float16* __restrict__ G) {
  const int lane = threadIdx.x & 31;
  const int wv   = threadIdx.x >> 5;
  const int gw   = blockIdx.x * 8 + wv;
  const int m0   = (gw >> 3) * 16;     // 256 M-tiles
  const int nb   = (gw & 7) * 128;     // 8 N-slabs of 128
  const int hi   = lane >> 4;
  const int l16  = lane & 15;

  v8f acc[8];
  #pragma unroll
  for (int t = 0; t < 8; ++t) { v8f z = {}; acc[t] = z; }

  const float* yrow = Y + (size_t)(m0 + l16) * DD;
  const _Float16* wbase = Wh + (size_t)(nb + l16) * DD;
  for (int k = 0; k < DD; k += 32) {
    // batch-issue all loads for this K step
    vf8 f0 = *(const vf8*)(yrow + k + 16 * hi);
    vf8 f1 = *(const vf8*)(yrow + k + 8 + 16 * hi);
    v16h bf[8];
    #pragma unroll
    for (int t = 0; t < 8; ++t)
      bf[t] = *(const v16h*)(wbase + (size_t)t * 16 * DD + k + 16 * hi);

    v8h a0, a1;
    #pragma unroll
    for (int i = 0; i < 8; ++i) { a0[i] = (_Float16)f0[i]; a1[i] = (_Float16)f1[i]; }
    v16h a = cat8(a0, a1);
    #pragma unroll
    for (int t = 0; t < 8; ++t) acc[t] = wmma16(a, bf[t], acc[t]);
  }
  // C layout: element r of v8f is row (r + 8*hi), column (l16) of the tile.
  #pragma unroll
  for (int t = 0; t < 8; ++t) {
    #pragma unroll
    for (int r = 0; r < 8; ++r) {
      G[(size_t)(m0 + r + 8 * hi) * DD + nb + t * 16 + l16] = (_Float16)acc[t][r];
    }
  }
}

// --------------------- fused flash attention -------------------------------
// One block = 16 query rows, 8 waves. Wave w owns output columns [w*128,w*128+128)
// and the K-slice [w*128, w*128+128) of the score dot-products.
// Per 32-key step: partial 16x32 score tile per wave -> deterministic LDS
// reduction (fixed wave order, no float atomics) -> online softmax (shfl row
// reductions) -> P through per-wave LDS (C-layout -> A-layout) -> 8 WMMAs
// accumulating P @ Y (B-frags are contiguous rows of Yt, preloaded in a batch).

__launch_bounds__(256)
__global__ void k_attn(const _Float16* __restrict__ G, const _Float16* __restrict__ Yt,
                       float* __restrict__ out) {
  __shared__ alignas(64) float    Sp[8][2][32][8];   // [wave][subtile][lane][r] 16KB
  __shared__ alignas(64) _Float16 Pb[8][16][32];     // per-wave P scratch      8KB

  const int lane = threadIdx.x & 31;
  const int wv   = threadIdx.x >> 5;
  const int hi   = lane >> 4;
  const int l16  = lane & 15;
  const int q0   = blockIdx.x * 16;
  const int cb   = wv * 128;    // output column slab
  const int kb   = wv * 128;    // score K-slice

  v8f acc[8];
  #pragma unroll
  for (int t = 0; t < 8; ++t) { v8f z = {}; acc[t] = z; }
  float mrow[8], lrow[8];
  #pragma unroll
  for (int r = 0; r < 8; ++r) { mrow[r] = -3.0e38f; lrow[r] = 0.0f; }

  const _Float16* qbase = G + (size_t)(q0 + l16) * DD;
  const _Float16* vbase = Yt + (size_t)(cb + l16) * NT;

  for (int j = 0; j < NT; j += 32) {
    // ---- partial scores over this wave's 128-wide K slice -----------------
    v8f s0 = {}, s1 = {};
    #pragma unroll
    for (int kk = 0; kk < 128; kk += 32) {
      const int k = kb + kk;
      v8h a0 = *(const v8h*)(qbase + k + 16 * hi);
      v8h a1 = *(const v8h*)(qbase + k + 8 + 16 * hi);
      // B[k][n] = G[j + n][k] : contiguous row reads of G
      v16h b0 = *(const v16h*)(G + (size_t)(j + l16) * DD + k + 16 * hi);
      v16h b1 = *(const v16h*)(G + (size_t)(j + 16 + l16) * DD + k + 16 * hi);
      v16h a = cat8(a0, a1);
      s0 = wmma16(a, b0, s0);
      s1 = wmma16(a, b1, s1);
    }
    *(v8f*)&Sp[wv][0][lane][0] = s0;
    *(v8f*)&Sp[wv][1][lane][0] = s1;
    __syncthreads();

    // ---- deterministic 8-way reduction (fixed order) ----------------------
    v8f t0 = {}, t1 = {};
    #pragma unroll
    for (int w = 0; w < 8; ++w) {
      t0 += *(const v8f*)&Sp[w][0][lane][0];
      t1 += *(const v8f*)&Sp[w][1][lane][0];
    }
    __syncthreads();   // reads consumed before next iteration overwrites Sp

    // ---- preload P@V B-fragments while softmax runs -----------------------
    // B[k][n] = Y[j+k][cb+n] = Yt[cb+n][j+k] : contiguous row reads of Yt
    v16h bY[8];
    #pragma unroll
    for (int t = 0; t < 8; ++t)
      bY[t] = *(const v16h*)(vbase + (size_t)t * 16 * NT + j + 16 * hi);

    // ---- online softmax ---------------------------------------------------
    float cscale[8];
    #pragma unroll
    for (int r = 0; r < 8; ++r) {
      float rm   = redmax16(fmaxf(t0[r], t1[r]));
      float mnew = fmaxf(mrow[r], rm);
      float sc   = __expf(mrow[r] - mnew);
      float p0   = __expf(t0[r] - mnew);
      float p1   = __expf(t1[r] - mnew);
      float rs   = redsum16(p0 + p1);
      lrow[r]    = lrow[r] * sc + rs;
      mrow[r]    = mnew;
      cscale[r]  = sc;
      // C-layout element (row r+8*hi, col l16 / 16+l16) -> row-major LDS
      Pb[wv][r + 8 * hi][l16]      = (_Float16)p0;
      Pb[wv][r + 8 * hi][16 + l16] = (_Float16)p1;
    }
    // per-wave LDS is serviced in-order; fence the compiler + split counter
    __asm__ volatile("s_wait_dscnt 0" ::: "memory");

    // ---- reload P in A-fragment layout ------------------------------------
    v8h pa0 = *(const v8h*)(&Pb[wv][l16][0] + 16 * hi);
    v8h pa1 = *(const v8h*)(&Pb[wv][l16][0] + 8 + 16 * hi);
    v16h pa = cat8(pa0, pa1);

    // ---- acc = acc*scale + P @ Y[j:j+32, cb:cb+128] -----------------------
    #pragma unroll
    for (int t = 0; t < 8; ++t) {
      #pragma unroll
      for (int r = 0; r < 8; ++r) acc[t][r] *= cscale[r];
      acc[t] = wmma16(pa, bY[t], acc[t]);
    }
  }

  // ---- normalize and store ------------------------------------------------
  #pragma unroll
  for (int t = 0; t < 8; ++t) {
    #pragma unroll
    for (int r = 0; r < 8; ++r) {
      out[(size_t)(q0 + r + 8 * hi) * DD + cb + t * 16 + l16] = acc[t][r] / lrow[r];
    }
  }
}

// ---------------------------------------------------------------------------

extern "C" void kernel_launch(void* const* d_in, const int* in_sizes, int n_in,
                              void* d_out, int out_size, void* d_ws, size_t ws_size,
                              hipStream_t stream) {
  (void)in_sizes; (void)n_in; (void)out_size; (void)ws_size;
  const float* Y = (const float*)d_in[0];       // (NT, DD) f32
  const float* W = (const float*)d_in[1];       // (DD, DD) f32
  float* out = (float*)d_out;                   // (NT, DD) f32

  char* ws = (char*)d_ws;                       // needs 18 MB
  _Float16* Wh = (_Float16*)ws;                                               // 2 MB
  _Float16* Yt = (_Float16*)(ws + (size_t)DD * DD * 2);                       // 8 MB
  _Float16* G  = (_Float16*)(ws + (size_t)DD * DD * 2 + (size_t)DD * NT * 2); // 8 MB

  k_cast_w     <<<(DD * DD) / 256,            256, 0, stream>>>(W, Wh);
  k_transpose_y<<<(DD * NT) / 256,            256, 0, stream>>>(Y, Yt);
  k_gemm1      <<<(NT / 16) * (DD / 128) / 8, 256, 0, stream>>>(Y, Wh, G);
  k_attn       <<<NT / 16,                    256, 0, stream>>>(G, Yt, out);
}
